// Model_25744033972954
// MI455X (gfx1250) — matrix operand order repaired and checked by reading, not compiled, not run on hardware
//
#include <hip/hip_runtime.h>
#include <hip/hip_bf16.h>
#include <math.h>

typedef __attribute__((ext_vector_type(16))) _Float16 v16h;
typedef __attribute__((ext_vector_type(8)))  float    v8f;
typedef __attribute__((ext_vector_type(2)))  _Float16 h2;

#define CDIV(a,b) (((a)+(b)-1)/(b))

// ============================================================================
// f32 -> f16 convert with zero padding into (rowsP x colsP) row-major buffer.
// Writes packed pairs (b32 stores). src has row stride ld.
// ============================================================================
__global__ void k_cvt_f16(const float* __restrict__ src, int rows, int cols, int ld,
                          _Float16* __restrict__ dst, int rowsP, int colsP)
{
  int t = blockIdx.x * blockDim.x + threadIdx.x;
  int total = rowsP * (colsP >> 1);
  if (t >= total) return;
  int cp2 = colsP >> 1;
  int c  = (t % cp2) * 2;
  int r  = t / cp2;
  _Float16 a = (_Float16)0.f, b = (_Float16)0.f;
  if (r < rows) {
    if (c < cols)     a = (_Float16)src[(size_t)r * ld + c];
    if (c + 1 < cols) b = (_Float16)src[(size_t)r * ld + c + 1];
  }
  h2 p; p[0] = a; p[1] = b;
  reinterpret_cast<h2*>(dst)[t] = p;
}

// ============================================================================
// WMMA GEMM on padded f16 operands:
//   C[M,O] = A16[Mp,Kp] * W16[Op,Kp]^T + bias   (f32 accumulate)
// Block: 256 threads = 8 waves; tile 64(M) x 64(O); K-step 32.
// Each wave computes two 16x16 tiles (shares B fragment) -> 2 WMMAs / K-step.
// Branch-free staging: b128 global loads -> b128 LDS stores.
// ============================================================================
__global__ __launch_bounds__(256)
void k_wmma_gemm16(const _Float16* __restrict__ A, const _Float16* __restrict__ W,
                   const float* __restrict__ bias, float* __restrict__ C,
                   int M, int Kp, int O, int relu)
{
  __shared__ _Float16 As[64][32];
  __shared__ _Float16 Bs[64][32];
  const int tid   = threadIdx.x;
  const int mBase = blockIdx.x * 64;
  const int nBase = blockIdx.y * 64;
  const int lane  = tid & 31;
  const int wave  = tid >> 5;
  const int waveM = wave >> 2;   // 0..1
  const int waveN = wave & 3;    // 0..3
  const int half  = lane >> 4;
  const int l16   = lane & 15;
  const int mLoc  = waveM * 16 + l16;
  const int nLoc  = waveN * 16 + l16;

  const int sRow = tid >> 2;          // 0..63
  const int sK   = (tid & 3) * 8;     // 0,8,16,24 (halves)

  v8f acc0 = {0.f,0.f,0.f,0.f,0.f,0.f,0.f,0.f};
  v8f acc1 = {0.f,0.f,0.f,0.f,0.f,0.f,0.f,0.f};

  for (int kBase = 0; kBase < Kp; kBase += 32) {
    // branch-free staging: rows are guaranteed in-bounds by padding
    const uint4 va = *reinterpret_cast<const uint4*>(
        &A[(size_t)(mBase + sRow) * Kp + kBase + sK]);
    const uint4 vb = *reinterpret_cast<const uint4*>(
        &W[(size_t)(nBase + sRow) * Kp + kBase + sK]);
    if (kBase + 32 < Kp) {
      __builtin_prefetch(&A[(size_t)(mBase + sRow) * Kp + kBase + 32 + sK], 0, 1);
      __builtin_prefetch(&W[(size_t)(nBase + sRow) * Kp + kBase + 32 + sK], 0, 1);
    }
    __syncthreads();   // previous iteration's fragments consumed
    *reinterpret_cast<uint4*>(&As[sRow][sK]) = va;
    *reinterpret_cast<uint4*>(&Bs[sRow][sK]) = vb;
    __syncthreads();

    v16h af0, af1, bf;
#pragma unroll
    for (int j = 0; j < 8; ++j) {
      // A 16x32 f16 layout (ISA 7.12.2): VGPR j<4 -> K = 8*half+2j ; j>=4 -> 16+8*half+2(j-4)
      int ka = (j < 4) ? (8 * half + 2 * j) : (16 + 8 * half + 2 * (j - 4));
      af0[2*j]   = As[mLoc][ka];
      af0[2*j+1] = As[mLoc][ka + 1];
      af1[2*j]   = As[mLoc + 32][ka];
      af1[2*j+1] = As[mLoc + 32][ka + 1];
      // B 32x16 f16 layout: same K order as A (j<4 -> K = 8*half+2j ; j>=4 -> 16+8*half+2(j-4))
      int kb = (j < 4) ? (8 * half + 2 * j) : (16 + 8 * half + 2 * (j - 4));
      bf[2*j]   = Bs[nLoc][kb];
      bf[2*j+1] = Bs[nLoc][kb + 1];
    }
    acc0 = __builtin_amdgcn_wmma_f32_16x16x32_f16(false, af0, false, bf,
                                                  (short)0, acc0, false, false);
    acc1 = __builtin_amdgcn_wmma_f32_16x16x32_f16(false, af1, false, bf,
                                                  (short)0, acc1, false, false);
  }

  const int gn = nBase + waveN * 16 + l16;   // C/D: N = lane % 16
  const float bv = (gn < O && bias) ? bias[gn] : 0.f;
#pragma unroll
  for (int r = 0; r < 8; ++r) {
    int gm0 = mBase + waveM * 16 + half * 8 + r;  // C/D: VGPR r -> M = r + 8*(lane/16)
    int gm1 = gm0 + 32;
    if (gm0 < M && gn < O) {
      float v = acc0[r] + bv;
      if (relu) v = fmaxf(v, 0.f);
      C[(size_t)gm0 * O + gn] = v;
    }
    if (gm1 < M && gn < O) {
      float v = acc1[r] + bv;
      if (relu) v = fmaxf(v, 0.f);
      C[(size_t)gm1 * O + gn] = v;
    }
  }
}

// ============================================================================
// Transpose x (B,C,N) -> pts (B,N,C), plus xyz0 = first 3 channels
// ============================================================================
__global__ void k_transpose_x(const float* __restrict__ x, float* __restrict__ pts,
                              float* __restrict__ xyz0, int B, int C, int N)
{
  int t = blockIdx.x * blockDim.x + threadIdx.x;
  int total = B * N * C;
  if (t >= total) return;
  int c = t % C;
  int bn = t / C;
  int n = bn % N;
  int b = bn / N;
  float v = x[((size_t)b * C + c) * N + n];
  pts[t] = v;
  if (c < 3) xyz0[(size_t)bn * 3 + c] = v;
}

// points1 for fp1: concat([xyz0, pts], -1) -> (B,N,27)
__global__ void k_p1cat(const float* __restrict__ x, float* __restrict__ out,
                        int B, int C, int N)
{
  int t = blockIdx.x * blockDim.x + threadIdx.x;
  int total = B * N * 27;
  if (t >= total) return;
  int c = t % 27;
  int bn = t / 27;
  int n = bn % N;
  int b = bn / N;
  int ch = (c < 3) ? c : (c - 3);
  out[t] = x[((size_t)b * C + ch) * N + n];
}

// ============================================================================
// Farthest point sampling: one workgroup per batch; sequential npoint iters
// with LDS dist array + argmax tree reduce (ties -> smallest index).
// ============================================================================
__global__ __launch_bounds__(256)
void k_fps(const float* __restrict__ xyz, int n, int npoint, int* __restrict__ idx)
{
  __shared__ float dist[2048];
  __shared__ float rmax[256];
  __shared__ int   rarg[256];
  const int b = blockIdx.x;
  const int tid = threadIdx.x;
  const float* p = xyz + (size_t)b * n * 3;
  for (int i = tid; i < n; i += 256) dist[i] = 1e10f;
  __syncthreads();
  int far = 0;
  for (int it = 0; it < npoint; ++it) {
    if (tid == 0) idx[b * npoint + it] = far;
    float cx = p[far * 3 + 0], cy = p[far * 3 + 1], cz = p[far * 3 + 2];
    float bm = -1.f;
    int   ba = 0;
    for (int i = tid; i < n; i += 256) {
      float dx = p[i * 3 + 0] - cx, dy = p[i * 3 + 1] - cy, dz = p[i * 3 + 2] - cz;
      float nd = fminf(dist[i], dx * dx + dy * dy + dz * dz);
      dist[i] = nd;
      if (nd > bm) { bm = nd; ba = i; }
    }
    rmax[tid] = bm; rarg[tid] = ba;
    __syncthreads();
    for (int s = 128; s > 0; s >>= 1) {
      if (tid < s) {
        float ov = rmax[tid + s]; int oa = rarg[tid + s];
        if (ov > rmax[tid] || (ov == rmax[tid] && oa < rarg[tid])) {
          rmax[tid] = ov; rarg[tid] = oa;
        }
      }
      __syncthreads();
    }
    far = rarg[0];
    __syncthreads();
  }
}

__global__ void k_gather_xyz(const float* __restrict__ xyz, const int* __restrict__ idx,
                             float* __restrict__ out, int n, int npoint, int B)
{
  int t = blockIdx.x * blockDim.x + threadIdx.x;
  int total = B * npoint * 3;
  if (t >= total) return;
  int d = t % 3;
  int s = (t / 3) % npoint;
  int b = t / (3 * npoint);
  int gi = idx[b * npoint + s];
  out[t] = xyz[((size_t)b * n + gi) * 3 + d];
}

// Ball query: first nsample indices (ascending) with d <= r^2; pad with first
// (or n-1 if none, matching jax gather clamp).
__global__ void k_ball_query(const float* __restrict__ xyz, const float* __restrict__ nxyz,
                             int* __restrict__ out, int n, int npoint, int nsample,
                             float r2, int B)
{
  int t = blockIdx.x * blockDim.x + threadIdx.x;
  if (t >= B * npoint) return;
  int b = t / npoint;
  const float* p = xyz + (size_t)b * n * 3;
  float qx = nxyz[(size_t)t * 3 + 0], qy = nxyz[(size_t)t * 3 + 1], qz = nxyz[(size_t)t * 3 + 2];
  int* o = out + (size_t)t * nsample;
  int cnt = 0;
  for (int i = 0; i < n && cnt < nsample; ++i) {
    float dx = p[i * 3] - qx, dy = p[i * 3 + 1] - qy, dz = p[i * 3 + 2] - qz;
    if (dx * dx + dy * dy + dz * dz <= r2) o[cnt++] = i;
  }
  int pad = (cnt > 0) ? o[0] : (n - 1);
  for (int k = cnt; k < nsample; ++k) o[k] = pad;
}

// Grouping: H[(b,s,nn), c] = c<3 ? xyz[gidx]-new_xyz : pts[gidx][c-3]
__global__ void k_group(const float* __restrict__ xyz, const float* __restrict__ pts,
                        const float* __restrict__ nxyz, const int* __restrict__ gidx,
                        float* __restrict__ H, int n, int npoint, int nsample,
                        int cpts, int B)
{
  int K = 3 + cpts;
  int t = blockIdx.x * blockDim.x + threadIdx.x;
  int total = B * npoint * nsample * K;
  if (t >= total) return;
  int c  = t % K;
  int r  = t / K;            // (b*npoint + s)*nsample + nn
  int nn = r % nsample;
  int sr = r / nsample;      // b*npoint + s
  int b  = sr / npoint;
  int gi = gidx[(size_t)sr * nsample + nn];
  float v;
  if (c < 3) v = xyz[((size_t)b * n + gi) * 3 + c] - nxyz[(size_t)sr * 3 + c];
  else       v = pts[((size_t)b * n + gi) * cpts + (c - 3)];
  H[t] = v;
}

// BN statistics per channel over all M rows (population variance)
__global__ __launch_bounds__(256)
void k_bn_stats(const float* __restrict__ H, int M, int O,
                float* __restrict__ mean, float* __restrict__ istd)
{
  __shared__ float s1[256], s2[256];
  int o = blockIdx.x, tid = threadIdx.x;
  float a = 0.f, b = 0.f;
  for (int r = tid; r < M; r += 256) {
    float v = H[(size_t)r * O + o];
    a += v; b += v * v;
  }
  s1[tid] = a; s2[tid] = b;
  __syncthreads();
  for (int s = 128; s > 0; s >>= 1) {
    if (tid < s) { s1[tid] += s1[tid + s]; s2[tid] += s2[tid + s]; }
    __syncthreads();
  }
  if (tid == 0) {
    float m = s1[0] / (float)M;
    float v = s2[0] / (float)M - m * m;
    mean[o] = m;
    istd[o] = rsqrtf(v + 1e-5f);
  }
}

__global__ void k_bn_relu(float* __restrict__ H, int total, int O,
                          const float* __restrict__ mean, const float* __restrict__ istd,
                          const float* __restrict__ g, const float* __restrict__ be)
{
  int t = blockIdx.x * blockDim.x + threadIdx.x;
  if (t >= total) return;
  int o = t % O;
  float v = (H[t] - mean[o]) * istd[o] * g[o] + be[o];
  H[t] = fmaxf(v, 0.f);
}

// max over nsample: out[(b,s),o] = max_nn H[((b,s)*nsample+nn), o]
__global__ void k_maxpool(const float* __restrict__ H, float* __restrict__ out,
                          int npoint, int nsample, int O, int B)
{
  int t = blockIdx.x * blockDim.x + threadIdx.x;
  int total = B * npoint * O;
  if (t >= total) return;
  int o = t % O;
  int sr = t / O;
  const float* p = H + ((size_t)sr * nsample) * O + o;
  float m = -1e30f;
  for (int k = 0; k < nsample; ++k) m = fmaxf(m, p[(size_t)k * O]);
  out[t] = m;
}

// 3 nearest neighbors + normalized inverse-distance weights
__global__ void k_three_nn(const float* __restrict__ xyz1, const float* __restrict__ xyz2,
                           int n1, int n2, int* __restrict__ idx, float* __restrict__ w, int B)
{
  int t = blockIdx.x * blockDim.x + threadIdx.x;
  if (t >= B * n1) return;
  int b = t / n1;
  float qx = xyz1[(size_t)t * 3], qy = xyz1[(size_t)t * 3 + 1], qz = xyz1[(size_t)t * 3 + 2];
  const float* p = xyz2 + (size_t)b * n2 * 3;
  float d0 = 1e30f, d1 = 1e30f, d2 = 1e30f;
  int i0 = 0, i1 = 0, i2 = 0;
  for (int i = 0; i < n2; ++i) {
    float dx = p[i * 3] - qx, dy = p[i * 3 + 1] - qy, dz = p[i * 3 + 2] - qz;
    float d = dx * dx + dy * dy + dz * dz;
    if (d < d0)      { d2 = d1; i2 = i1; d1 = d0; i1 = i0; d0 = d; i0 = i; }
    else if (d < d1) { d2 = d1; i2 = i1; d1 = d;  i1 = i; }
    else if (d < d2) { d2 = d;  i2 = i; }
  }
  float w0 = 1.f / (d0 + 1e-8f), w1 = 1.f / (d1 + 1e-8f), w2 = 1.f / (d2 + 1e-8f);
  float s = w0 + w1 + w2;
  idx[(size_t)t * 3 + 0] = i0; idx[(size_t)t * 3 + 1] = i1; idx[(size_t)t * 3 + 2] = i2;
  w[(size_t)t * 3 + 0] = w0 / s; w[(size_t)t * 3 + 1] = w1 / s; w[(size_t)t * 3 + 2] = w2 / s;
}

// concat([points1, interp(points2)], -1)
__global__ void k_fp_concat(const float* __restrict__ p1, int c1,
                            const int* __restrict__ idx, const float* __restrict__ w,
                            const float* __restrict__ p2, int c2, int n2,
                            float* __restrict__ H, int n1, int B)
{
  int K = c1 + c2;
  int t = blockIdx.x * blockDim.x + threadIdx.x;
  int total = B * n1 * K;
  if (t >= total) return;
  int c = t % K;
  int r = t / K;        // b*n1 + n
  int b = r / n1;
  float v;
  if (c < c1) v = p1[(size_t)r * c1 + c];
  else {
    int cc = c - c1;
    const float* pp = p2 + (size_t)b * n2 * c2;
    v = w[(size_t)r * 3 + 0] * pp[(size_t)idx[(size_t)r * 3 + 0] * c2 + cc]
      + w[(size_t)r * 3 + 1] * pp[(size_t)idx[(size_t)r * 3 + 1] * c2 + cc]
      + w[(size_t)r * 3 + 2] * pp[(size_t)idx[(size_t)r * 3 + 2] * c2 + cc];
  }
  H[t] = v;
}

// global max over rows per (b, c)
__global__ void k_global_max(const float* __restrict__ H, float* __restrict__ out,
                             int rows, int C, int B)
{
  int t = blockIdx.x * blockDim.x + threadIdx.x;
  if (t >= B * C) return;
  int c = t % C;
  int b = t / C;
  const float* p = H + ((size_t)b * rows) * C + c;
  float m = -1e30f;
  for (int r = 0; r < rows; ++r) m = fmaxf(m, p[(size_t)r * C]);
  out[t] = m;
}

// conv1 decomposed: out = relu(bias + gvec[b] + sum_{c<5} feat5[m,c]*W1[o,c])
// feat5 = [grid(2), center(3)]; grid from n%16, center from fc3 output h3.
__global__ void k_conv1_fused(const float* __restrict__ h3, const float* __restrict__ gvec,
                              const float* __restrict__ W1, const float* __restrict__ b1,
                              float* __restrict__ out, int B, int N)
{
  int t = blockIdx.x * blockDim.x + threadIdx.x;
  int total = B * N * 512;
  if (t >= total) return;
  int o = t % 512;
  int m = t / 512;
  int n = m % N;
  int b = m / N;
  const float gx[4] = {-0.05f, -0.016666668f, 0.016666668f, 0.05f};
  int k = n & 15;
  float f0 = gx[k >> 2];
  float f1 = gx[k & 3];
  int cs = n >> 4;
  float c0 = h3[b * 3072 + cs];
  float c1 = h3[b * 3072 + 1024 + cs];
  float c2 = h3[b * 3072 + 2048 + cs];
  const float* wr = W1 + (size_t)o * 1029;
  float v = b1[o] + gvec[b * 512 + o]
          + f0 * wr[0] + f1 * wr[1] + c0 * wr[2] + c1 * wr[3] + c2 * wr[4];
  out[t] = fmaxf(v, 0.f);
}

// out2 = conv3 + center (writes straight into d_out, row-major (b,n,3))
__global__ void k_fine_out(const float* __restrict__ c3out, const float* __restrict__ h3,
                           float* __restrict__ out, int B, int N)
{
  int t = blockIdx.x * blockDim.x + threadIdx.x;
  int total = B * N * 3;
  if (t >= total) return;
  int d = t % 3;
  int n = (t / 3) % N;
  int b = t / (3 * N);
  out[t] = c3out[t] + h3[b * 3072 + d * 1024 + (n >> 4)];
}

// out1 (b,1024,3) from fc3 output
__global__ void k_out1(const float* __restrict__ h3, float* __restrict__ out, int B)
{
  int t = blockIdx.x * blockDim.x + threadIdx.x;
  int total = B * 1024 * 3;
  if (t >= total) return;
  int d = t % 3;
  int s = (t / 3) % 1024;
  int b = t / 3072;
  out[t] = h3[b * 3072 + d * 1024 + s];
}

// brute-force NN min distance with LDS tiling of target cloud; stores sqrt(min d2)
__global__ __launch_bounds__(256)
void k_min_dist(const float* __restrict__ Aq, int na, const float* __restrict__ Bp, int nb,
                float* __restrict__ out)
{
  __shared__ float sx[256], sy[256], sz[256];
  int b = blockIdx.y;
  int q = blockIdx.x * 256 + threadIdx.x;
  float qx = 0.f, qy = 0.f, qz = 0.f;
  if (q < na) {
    qx = Aq[((size_t)b * na + q) * 3 + 0];
    qy = Aq[((size_t)b * na + q) * 3 + 1];
    qz = Aq[((size_t)b * na + q) * 3 + 2];
  }
  float best = 1e30f;
  for (int t0 = 0; t0 < nb; t0 += 256) {
    int i = t0 + threadIdx.x;
    if (i < nb) {
      sx[threadIdx.x] = Bp[((size_t)b * nb + i) * 3 + 0];
      sy[threadIdx.x] = Bp[((size_t)b * nb + i) * 3 + 1];
      sz[threadIdx.x] = Bp[((size_t)b * nb + i) * 3 + 2];
    }
    __syncthreads();
    int cnt = min(256, nb - t0);
    for (int j = 0; j < cnt; ++j) {
      float dx = qx - sx[j], dy = qy - sy[j], dz = qz - sz[j];
      best = fminf(best, dx * dx + dy * dy + dz * dz);
    }
    __syncthreads();
  }
  if (q < na) out[(size_t)b * na + q] = sqrtf(best);
}

__global__ __launch_bounds__(256)
void k_cd_mean(const float* __restrict__ d1, int n1, const float* __restrict__ d2, int n2,
               float* __restrict__ loss)
{
  __shared__ float s1[256], s2[256];
  int b = blockIdx.x, tid = threadIdx.x;
  float a = 0.f, c = 0.f;
  for (int i = tid; i < n1; i += 256) a += d1[(size_t)b * n1 + i];
  for (int i = tid; i < n2; i += 256) c += d2[(size_t)b * n2 + i];
  s1[tid] = a; s2[tid] = c;
  __syncthreads();
  for (int s = 128; s > 0; s >>= 1) {
    if (tid < s) { s1[tid] += s1[tid + s]; s2[tid] += s2[tid + s]; }
    __syncthreads();
  }
  if (tid == 0) loss[b] = 0.5f * (s1[0] / (float)n1 + s2[0] / (float)n2);
}

__global__ void k_finalize(const float* __restrict__ l1, const float* __restrict__ l2,
                           float* __restrict__ outT)
{
  if (blockIdx.x == 0 && threadIdx.x == 0) {
    outT[0] = l2[0];
    outT[1] = l2[1];
    outT[2] = 0.5f * (l1[0] + l1[1]) + 0.25f * (l2[0] + l2[1]);  // mean(l1) + 0.5*mean(l2)
  }
}

// ============================================================================
extern "C" void kernel_launch(void* const* d_in, const int* in_sizes, int n_in,
                              void* d_out, int out_size, void* d_ws, size_t ws_size,
                              hipStream_t stream) {
  (void)in_sizes; (void)n_in; (void)out_size; (void)ws_size;
  const int B = 2, CIN = 24, NIN = 2048;
  const int NP1 = 1024, NS1 = 16, NP2 = 512, NS2 = 32;
  const int NPTS = 16384, NGT = 16384;

  const float* x  = (const float*)d_in[0];
  const float* gt = (const float*)d_in[1];
  // params flattened: sorted keys, layer dicts as {b, be, g, w}
  const float* conv1_b = (const float*)d_in[2];
  const float* conv1_w = (const float*)d_in[3];
  const float* conv2_b = (const float*)d_in[4];
  const float* conv2_w = (const float*)d_in[5];
  const float* conv3_b = (const float*)d_in[6];
  const float* conv3_w = (const float*)d_in[7];
  const float* fc1_b = (const float*)d_in[8];
  const float* fc1_w = (const float*)d_in[9];
  const float* fc2_b = (const float*)d_in[10];
  const float* fc2_w = (const float*)d_in[11];
  const float* fc3_b = (const float*)d_in[12];
  const float* fc3_w = (const float*)d_in[13];
  const float* fp1_b[2]  = {(const float*)d_in[14], (const float*)d_in[18]};
  const float* fp1_be[2] = {(const float*)d_in[15], (const float*)d_in[19]};
  const float* fp1_g[2]  = {(const float*)d_in[16], (const float*)d_in[20]};
  const float* fp1_w[2]  = {(const float*)d_in[17], (const float*)d_in[21]};
  const float* fp2_b[2]  = {(const float*)d_in[22], (const float*)d_in[26]};
  const float* fp2_be[2] = {(const float*)d_in[23], (const float*)d_in[27]};
  const float* fp2_g[2]  = {(const float*)d_in[24], (const float*)d_in[28]};
  const float* fp2_w[2]  = {(const float*)d_in[25], (const float*)d_in[29]};
  const float* proj_b = (const float*)d_in[30];
  const float* proj_w = (const float*)d_in[31];
  const float* sa1_b[3]  = {(const float*)d_in[32], (const float*)d_in[36], (const float*)d_in[40]};
  const float* sa1_be[3] = {(const float*)d_in[33], (const float*)d_in[37], (const float*)d_in[41]};
  const float* sa1_g[3]  = {(const float*)d_in[34], (const float*)d_in[38], (const float*)d_in[42]};
  const float* sa1_w[3]  = {(const float*)d_in[35], (const float*)d_in[39], (const float*)d_in[43]};
  const float* sa2_b[2]  = {(const float*)d_in[44], (const float*)d_in[48]};
  const float* sa2_be[2] = {(const float*)d_in[45], (const float*)d_in[49]};
  const float* sa2_g[2]  = {(const float*)d_in[46], (const float*)d_in[50]};
  const float* sa2_w[2]  = {(const float*)d_in[47], (const float*)d_in[51]};

  float* outp = (float*)d_out;

  // ---- workspace carving (floats, 64-elt aligned) ----
  float* ws = (float*)d_ws;
  size_t off = 0;
  auto alloc = [&](size_t nelts) {
    float* p = ws + off;
    off += (nelts + 63) & ~(size_t)63;
    return p;
  };
  float* pts    = alloc((size_t)B * NIN * CIN);
  float* xyz0   = alloc((size_t)B * NIN * 3);
  int*   fidx1  = (int*)alloc((size_t)B * NP1);
  float* l1xyz  = alloc((size_t)B * NP1 * 3);
  int*   gidx1  = (int*)alloc((size_t)B * NP1 * NS1);
  int*   fidx2  = (int*)alloc((size_t)B * NP2);
  float* l2xyz  = alloc((size_t)B * NP2 * 3);
  int*   gidx2  = (int*)alloc((size_t)B * NP2 * NS2);
  float* l1pts  = alloc((size_t)B * NP1 * 128);
  float* l2pts  = alloc((size_t)B * NP2 * 256);
  float* l1pts2 = alloc((size_t)B * NP1 * 256);
  float* p1cat  = alloc((size_t)B * NIN * 27);
  int*   nnidx  = (int*)alloc((size_t)B * NIN * 3);
  float* nnw    = alloc((size_t)B * NIN * 3);
  float* gmax   = alloc((size_t)B * 256);
  float* enc    = alloc((size_t)B * 1024);
  float* h1     = alloc((size_t)B * 1024);
  float* h2v    = alloc((size_t)B * 1024);
  float* h3     = alloc((size_t)B * 3072);
  float* gvec   = alloc((size_t)B * 512);
  float* bnm    = alloc(512);
  float* bns    = alloc(512);
  float* out1   = alloc((size_t)B * NP1 * 3);
  float* c3out  = alloc((size_t)B * NPTS * 3);
  float* d1c    = alloc((size_t)B * NP1);
  float* d2c    = alloc((size_t)B * NGT);
  float* d1f    = alloc((size_t)B * NPTS);
  float* d2f    = alloc((size_t)B * NGT);
  float* loss1  = alloc(2);
  float* loss2  = alloc(2);
  float* bufA   = alloc((size_t)32768 * 512);
  float* bufB   = alloc((size_t)32768 * 512);
  // padded f16 staging buffers (allocated in float units, 2 halves per float)
  _Float16* A16 = (_Float16*)alloc((size_t)32768 * 512 / 2);  // up to 32768 x 512 halves
  _Float16* W16 = (_Float16*)alloc((size_t)3072 * 1024 / 2);  // up to 3072 x 1024 halves

  // GEMM driver: convert both operands to padded f16, then branch-free WMMA GEMM.
  auto gemm = [&](const float* A, int lda, const float* W, int ldw, const float* bias,
                  float* C, int M, int K, int O, int relu) {
    int Mp = CDIV(M, 64) * 64;
    int Kp = CDIV(K, 32) * 32;
    int Op = CDIV(O, 64) * 64;
    k_cvt_f16<<<CDIV(Mp * (Kp / 2), 256), 256, 0, stream>>>(A, M, K, lda, A16, Mp, Kp);
    k_cvt_f16<<<CDIV(Op * (Kp / 2), 256), 256, 0, stream>>>(W, O, K, ldw, W16, Op, Kp);
    dim3 g(Mp / 64, Op / 64);
    k_wmma_gemm16<<<g, 256, 0, stream>>>(A16, W16, bias, C, M, Kp, O, relu);
  };
  auto bn = [&](float* H, int M, int O, const float* g_, const float* be_) {
    k_bn_stats<<<O, 256, 0, stream>>>(H, M, O, bnm, bns);
    int total = M * O;
    k_bn_relu<<<CDIV(total, 256), 256, 0, stream>>>(H, total, O, bnm, bns, g_, be_);
  };

  // ================= encoder =================
  k_transpose_x<<<CDIV(B * NIN * CIN, 256), 256, 0, stream>>>(x, pts, xyz0, B, CIN, NIN);

  // --- SA1 ---
  k_fps<<<B, 256, 0, stream>>>(xyz0, NIN, NP1, fidx1);
  k_gather_xyz<<<CDIV(B * NP1 * 3, 256), 256, 0, stream>>>(xyz0, fidx1, l1xyz, NIN, NP1, B);
  k_ball_query<<<CDIV(B * NP1, 256), 256, 0, stream>>>(xyz0, l1xyz, gidx1, NIN, NP1, NS1,
                                                       0.035f * 0.035f, B);
  const int M1 = B * NP1 * NS1;  // 32768
  k_group<<<CDIV(M1 * 27, 256), 256, 0, stream>>>(xyz0, pts, l1xyz, gidx1, bufA,
                                                  NIN, NP1, NS1, CIN, B);
  gemm(bufA, 27, sa1_w[0], 27, sa1_b[0], bufB, M1, 27, 64, 0);     bn(bufB, M1, 64,  sa1_g[0], sa1_be[0]);
  gemm(bufB, 64, sa1_w[1], 64, sa1_b[1], bufA, M1, 64, 128, 0);    bn(bufA, M1, 128, sa1_g[1], sa1_be[1]);
  gemm(bufA, 128, sa1_w[2], 128, sa1_b[2], bufB, M1, 128, 128, 0); bn(bufB, M1, 128, sa1_g[2], sa1_be[2]);
  k_maxpool<<<CDIV(B * NP1 * 128, 256), 256, 0, stream>>>(bufB, l1pts, NP1, NS1, 128, B);

  // --- SA2 ---
  k_fps<<<B, 256, 0, stream>>>(l1xyz, NP1, NP2, fidx2);
  k_gather_xyz<<<CDIV(B * NP2 * 3, 256), 256, 0, stream>>>(l1xyz, fidx2, l2xyz, NP1, NP2, B);
  k_ball_query<<<CDIV(B * NP2, 256), 256, 0, stream>>>(l1xyz, l2xyz, gidx2, NP1, NP2, NS2,
                                                       0.0176f * 0.0176f, B);
  const int M2 = B * NP2 * NS2;  // 32768
  k_group<<<CDIV(M2 * 131, 256), 256, 0, stream>>>(l1xyz, l1pts, l2xyz, gidx2, bufA,
                                                   NP1, NP2, NS2, 128, B);
  gemm(bufA, 131, sa2_w[0], 131, sa2_b[0], bufB, M2, 131, 128, 0); bn(bufB, M2, 128, sa2_g[0], sa2_be[0]);
  gemm(bufB, 128, sa2_w[1], 128, sa2_b[1], bufA, M2, 128, 256, 0); bn(bufA, M2, 256, sa2_g[1], sa2_be[1]);
  k_maxpool<<<CDIV(B * NP2 * 256, 256), 256, 0, stream>>>(bufA, l2pts, NP2, NS2, 256, B);

  // --- FP2 ---
  k_three_nn<<<CDIV(B * NP1, 256), 256, 0, stream>>>(l1xyz, l2xyz, NP1, NP2, nnidx, nnw, B);
  k_fp_concat<<<CDIV(B * NP1 * 384, 256), 256, 0, stream>>>(l1pts, 128, nnidx, nnw,
                                                            l2pts, 256, NP2, bufB, NP1, B);
  const int Mfp2 = B * NP1;  // 2048
  gemm(bufB, 384, fp2_w[0], 384, fp2_b[0], bufA, Mfp2, 384, 256, 0);   bn(bufA, Mfp2, 256, fp2_g[0], fp2_be[0]);
  gemm(bufA, 256, fp2_w[1], 256, fp2_b[1], l1pts2, Mfp2, 256, 256, 0); bn(l1pts2, Mfp2, 256, fp2_g[1], fp2_be[1]);

  // --- FP1 ---
  k_p1cat<<<CDIV(B * NIN * 27, 256), 256, 0, stream>>>(x, p1cat, B, CIN, NIN);
  k_three_nn<<<CDIV(B * NIN, 256), 256, 0, stream>>>(xyz0, l1xyz, NIN, NP1, nnidx, nnw, B);
  k_fp_concat<<<CDIV(B * NIN * 283, 256), 256, 0, stream>>>(p1cat, 27, nnidx, nnw,
                                                            l1pts2, 256, NP1, bufA, NIN, B);
  const int Mfp1 = B * NIN;  // 4096
  gemm(bufA, 283, fp1_w[0], 283, fp1_b[0], bufB, Mfp1, 283, 256, 0); bn(bufB, Mfp1, 256, fp1_g[0], fp1_be[0]);
  gemm(bufB, 256, fp1_w[1], 256, fp1_b[1], bufA, Mfp1, 256, 256, 0); bn(bufA, Mfp1, 256, fp1_g[1], fp1_be[1]);

  k_global_max<<<CDIV(B * 256, 256), 256, 0, stream>>>(bufA, gmax, NIN, 256, B);
  gemm(gmax, 256, proj_w, 256, proj_b, enc, B, 256, 1024, 0);

  // ================= decoder =================
  gemm(enc, 1024, fc1_w, 1024, fc1_b, h1, B, 1024, 1024, 1);
  gemm(h1, 1024, fc2_w, 1024, fc2_b, h2v, B, 1024, 1024, 1);
  gemm(h2v, 1024, fc3_w, 1024, fc3_b, h3, B, 1024, 3072, 0);
  // conv1 split: global-feature part is batch-constant -> tiny GEMM + rank-5 fixup
  gemm(enc, 1024, conv1_w + 5, 1029, nullptr, gvec, B, 1024, 512, 0);
  k_conv1_fused<<<CDIV(B * NPTS * 512, 256), 256, 0, stream>>>(h3, gvec, conv1_w, conv1_b,
                                                               bufA, B, NPTS);
  const int Mc = B * NPTS;  // 32768
  gemm(bufA, 512, conv2_w, 512, conv2_b, bufB, Mc, 512, 512, 1);
  gemm(bufB, 512, conv3_w, 512, conv3_b, c3out, Mc, 512, 3, 0);
  k_fine_out<<<CDIV(B * NPTS * 3, 256), 256, 0, stream>>>(c3out, h3, outp, B, NPTS);
  k_out1<<<CDIV(B * NP1 * 3, 256), 256, 0, stream>>>(h3, out1, B);

  // ================= chamfer =================
  k_min_dist<<<dim3(CDIV(NP1, 256), B), 256, 0, stream>>>(out1, NP1, gt, NGT, d1c);
  k_min_dist<<<dim3(CDIV(NGT, 256), B), 256, 0, stream>>>(gt, NGT, out1, NP1, d2c);
  k_min_dist<<<dim3(CDIV(NPTS, 256), B), 256, 0, stream>>>(outp, NPTS, gt, NGT, d1f);
  k_min_dist<<<dim3(CDIV(NGT, 256), B), 256, 0, stream>>>(gt, NGT, outp, NPTS, d2f);
  k_cd_mean<<<B, 256, 0, stream>>>(d1c, NP1, d2c, NGT, loss1);
  k_cd_mean<<<B, 256, 0, stream>>>(d1f, NPTS, d2f, NGT, loss2);
  k_finalize<<<1, 1, 0, stream>>>(loss1, loss2, outp + (size_t)B * NPTS * 3);
}
